// ViT_32255204393066
// MI455X (gfx1250) — compile-verified, matrix operand order checked by reading, not run on hardware
//
#include <hip/hip_runtime.h>
#include <math.h>

// ---------------------------------------------------------------------------
// ViT forward for MI455X (gfx1250, wave32). All GEMMs via V_WMMA_F32_16X16X32_BF16.
// Bandwidth-bound: images (154MB) read once; attention matrix never hits HBM
// (fused scores->softmax->att@V per 16-row strip via LDS). Weights pre-packed
// into the WMMA B-matrix lane layout (one 32B load per B tile); V stored
// transposed so att@V B tiles vectorize; bf16 conversion via native fptrunc
// (lets the backend pick v_cvt_pk_bf16_f32 on gfx1250).
// ---------------------------------------------------------------------------

typedef __attribute__((ext_vector_type(16))) __bf16 v16bf;
typedef __attribute__((ext_vector_type(8)))  __bf16 v8bf;
typedef __attribute__((ext_vector_type(2)))  __bf16 v2bf;
typedef __attribute__((ext_vector_type(8)))  float  v8f;

#define DEV static __device__ __forceinline__

constexpr int BATCH  = 256;
constexpr int SEQ    = 197;      // 14*14 patches + cls
constexpr int SEQP   = 224;      // padded to multiple of 32 (7 K-steps of 32)
constexpr int DIM    = 64;
constexpr int NHEADS = 2;
constexpr int DHEAD  = 32;
constexpr int NPATCH = 196;
constexpr int DPATCH = 768;      // 3*16*16
constexpr int NCLS   = 1000;
constexpr int NCLSP  = 1008;     // padded to multiple of 16
constexpr int NLAYER = 2;
constexpr float LNEPS = 1e-5f;

// Native conversions: fptrunc f32->bf16 is RNE; backend can select hardware
// cvt (v_cvt_pk_bf16_f32) on gfx1250 instead of the 3-op manual sequence.
DEV __bf16 f2bf(float f) { return (__bf16)f; }

DEV unsigned f2bf_pk(float a, float b) {
  v2bf r; r[0] = (__bf16)a; r[1] = (__bf16)b;
  return __builtin_bit_cast(unsigned, r);
}
DEV __bf16 bf_lo(unsigned u) { return __builtin_bit_cast(v2bf, u)[0]; }
DEV __bf16 bf_hi(unsigned u) { return __builtin_bit_cast(v2bf, u)[1]; }

// K-offset of element e (0..15) within a 32-wide K tile for lane-half h,
// per the CDNA5 16-bit A/B VGPR layout (ISA 7.12.2).
DEV int kmap(int e, int h) { return (e < 8) ? (h * 8 + e) : (16 + h * 8 + (e - 8)); }

// Load a 16x32 bf16 tile (A layout; also B^T layout) from row-major memory.
// Leading dim must be a multiple of 8 and kbase a multiple of 32 -> 16B aligned.
DEV v16bf load_tile16(const __bf16* __restrict__ A, int ld, int rbase, int kbase, int lane) {
  const int row = rbase + (lane & 15);
  const int h   = lane >> 4;
  const __bf16* p = A + (size_t)row * ld + kbase;
  v8bf lo = *(const v8bf*)(p + h * 8);
  v8bf hi = *(const v8bf*)(p + 16 + h * 8);
  v16bf r;
#pragma unroll
  for (int e = 0; e < 8; ++e) { r[e] = lo[e]; r[8 + e] = hi[e]; }
  return r;
}

// Load a pre-packed B tile: one 32-byte contiguous load per lane.
// Packed layout: dst[((kt*NT + nt)*32 + lane)*16 + e] = W[(kt*32+kmap(e,lane>>4))*N + nt*16 + (lane&15)]
DEV v16bf load_b_packed(const __bf16* __restrict__ Bp, int NT, int kt, int nt, int lane) {
  return *((const v16bf*)Bp + ((size_t)(kt * NT + nt) * 32 + lane));
}

#define WMMA(acc, a, b)                                                         \
  (acc) = __builtin_amdgcn_wmma_f32_16x16x32_bf16(false, (a), false, (b),       \
                                                  (short)0, (acc), false, false)

// ------------------------------- kernels -----------------------------------

// Pack nmat row-major fp32 (K x Nsrc) matrices into tiled WMMA-B bf16 layout
// with N (>=Nsrc, multiple of 16) columns; columns >= Nsrc are zero-filled.
__global__ void vit_pack_b(const float* __restrict__ src, __bf16* __restrict__ dst,
                           int K, int N, int Nsrc, int nmat) {
  int o = blockIdx.x * blockDim.x + threadIdx.x;
  int per = K * N;
  if (o >= per * nmat) return;
  int m    = o / per;
  int r    = o - m * per;
  int e    = r & 15;
  int lane = (r >> 4) & 31;
  int tile = r >> 9;                 // 512 elements per 32x16 tile
  int NT   = N >> 4;
  int nt   = tile % NT, kt = tile / NT;
  int kq   = kt * 32 + kmap(e, lane >> 4);
  int n    = nt * 16 + (lane & 15);
  dst[o] = (n < Nsrc) ? f2bf(src[(size_t)m * K * Nsrc + (size_t)kq * Nsrc + n]) : (__bf16)0.f;
}

// x[b,0,:] = cls + pos[0]; zero the padded rows 197..223.
__global__ void vit_init_rows(const float* __restrict__ cls, const float* __restrict__ pos,
                              float* __restrict__ xA) {
  const int b = blockIdx.x;
  const int t = threadIdx.x;                  // 64 threads
  float* xb = xA + (size_t)b * SEQP * DIM;
  xb[t] = cls[t] + pos[t];
  for (int i = t; i < (SEQP - SEQ) * DIM; i += 64)
    xb[(size_t)SEQ * DIM + i] = 0.f;
}

// Patch-embed GEMM: (B*196, 768) x (768, 64). One wave per 16x64 tile.
// A tile gathered directly from images: loop-invariant lane base pointer +
// compile-time offsets (c*50176 + step*448 (+224)) -> immediate-offset b128
// loads; packed bf16 conversion.
__global__ void vit_patch_embed(const float* __restrict__ images, const __bf16* __restrict__ Wpb,
                                const float* __restrict__ bpatch, const float* __restrict__ pos,
                                float* __restrict__ xA) {
  const int lane  = threadIdx.x;
  const int mbase = blockIdx.x * 16;
  const int h     = lane >> 4;
  v8f acc[4] = {};

  const int g  = mbase + (lane & 15);
  const int b  = g / NPATCH;
  const int p  = g - b * NPATCH;
  const int pi = p / 14, pj = p - pi * 14;
  // lane base: image(b), row pi*16, col pj*16 + h*8 (run start within a row)
  const float* base = images + (size_t)b * 3 * 224 * 224 + (size_t)(pi * 16) * 224 + pj * 16 + h * 8;

#pragma unroll
  for (int c = 0; c < 3; ++c) {
#pragma unroll
    for (int st = 0; st < 8; ++st) {
      const int kt = c * 8 + st;                       // K tile (k = kt*32)
      const float* pf = base + c * (224 * 224) + st * 448;   // rows 2*st, 2*st+1
      float4 f0 = *(const float4*)pf;                  // k = kt*32 + h*8 + 0..3
      float4 f1 = *(const float4*)(pf + 4);            // .. + 4..7
      float4 f2 = *(const float4*)(pf + 224);          // k = kt*32 + 16 + h*8 + 0..3
      float4 f3 = *(const float4*)(pf + 228);          // .. + 4..7
      union { v16bf v; unsigned u[8]; } a;
      a.u[0] = f2bf_pk(f0.x, f0.y);
      a.u[1] = f2bf_pk(f0.z, f0.w);
      a.u[2] = f2bf_pk(f1.x, f1.y);
      a.u[3] = f2bf_pk(f1.z, f1.w);
      a.u[4] = f2bf_pk(f2.x, f2.y);
      a.u[5] = f2bf_pk(f2.z, f2.w);
      a.u[6] = f2bf_pk(f3.x, f3.y);
      a.u[7] = f2bf_pk(f3.z, f3.w);
#pragma unroll
      for (int t = 0; t < 4; ++t) {
        v16bf bt = load_b_packed(Wpb, 4, kt, t, lane);
        WMMA(acc[t], a.v, bt);
      }
    }
  }
  const int col = lane & 15;
  // incremental (b, p) tracking instead of per-row division
  int gr0 = mbase + 8 * h;
  int bb  = gr0 / NPATCH;
  int pp  = gr0 - bb * NPATCH;
#pragma unroll
  for (int r = 0; r < 8; ++r) {
    int s = pp + 1;
    float* xrow = xA + ((size_t)bb * SEQP + s) * DIM;
    const float* prow = pos + (size_t)s * DIM;
#pragma unroll
    for (int t = 0; t < 4; ++t) {
      int n = t * 16 + col;
      xrow[n] = acc[t][r] + bpatch[n] + prow[n];
    }
    if (++pp == NPATCH) { pp = 0; ++bb; }
  }
}

// LayerNorm over D=64 per row; one wave per row; padded rows -> zeros (keeps K/V clean).
__global__ void vit_layernorm(const float* __restrict__ x, const float* __restrict__ g,
                              const float* __restrict__ bb, __bf16* __restrict__ y) {
  const int lane = threadIdx.x;
  const int row  = blockIdx.x;
  const int s    = row % SEQP;
  __bf16* yr = y + (size_t)row * DIM;
  if (s >= SEQ) { yr[lane] = (__bf16)0.f; yr[lane + 32] = (__bf16)0.f; return; }
  const float* xr = x + (size_t)row * DIM;
  float a0 = xr[lane], a1 = xr[lane + 32];
  float sum = a0 + a1;
#pragma unroll
  for (int o = 16; o >= 1; o >>= 1) sum += __shfl_xor(sum, o, 32);
  float mean = sum * (1.f / 64.f);
  float d0 = a0 - mean, d1 = a1 - mean;
  float vs = d0 * d0 + d1 * d1;
#pragma unroll
  for (int o = 16; o >= 1; o >>= 1) vs += __shfl_xor(vs, o, 32);
  float inv = rsqrtf(vs * (1.f / 64.f) + LNEPS);
  unsigned u = f2bf_pk(d0 * inv * g[lane] + bb[lane], d1 * inv * g[lane + 32] + bb[lane + 32]);
  yr[lane]      = bf_lo(u);
  yr[lane + 32] = bf_hi(u);
}

// QKV: per (b, head, 16-row tile): one A load, 6 WMMAs (q/k/v x two 16-col tiles).
// Q,K stored (bh, s, d); V stored TRANSPOSED (bh, d, s) so the att@V B tiles
// become contiguous-K vector loads.
__global__ void vit_qkv(const __bf16* __restrict__ y,
                        const __bf16* __restrict__ qWb, const __bf16* __restrict__ kWb,
                        const __bf16* __restrict__ vWb,
                        const float* __restrict__ qb, const float* __restrict__ kb,
                        const float* __restrict__ vb,
                        __bf16* __restrict__ q, __bf16* __restrict__ k, __bf16* __restrict__ vT) {
  const int lane = threadIdx.x;
  int idx = blockIdx.x;
  const int mt = idx % (SEQP / 16); idx /= (SEQP / 16);
  const int h  = idx % NHEADS;      idx /= NHEADS;
  const int b  = idx;
  const int bh = b * NHEADS + h;

  v16bf a = load_tile16(y + (size_t)b * SEQP * DIM, DIM, mt * 16, h * DHEAD, lane);

  const __bf16* Ws[3] = { qWb + h * DHEAD * DHEAD, kWb + h * DHEAD * DHEAD, vWb + h * DHEAD * DHEAD };
  const float*  bs[3] = { qb + h * DHEAD, kb + h * DHEAD, vb + h * DHEAD };

  const int col = lane & 15;
#pragma unroll
  for (int m = 0; m < 3; ++m) {
    v8f acc0 = {}, acc1 = {};
    v16bf b0 = load_b_packed(Ws[m], 2, 0, 0, lane);
    v16bf b1 = load_b_packed(Ws[m], 2, 0, 1, lane);
    WMMA(acc0, a, b0);
    WMMA(acc1, a, b1);
    float bias0 = bs[m][col], bias1 = bs[m][16 + col];
#pragma unroll
    for (int r = 0; r < 8; ++r) {
      int srow = mt * 16 + r + 8 * (lane >> 4);
      unsigned u = f2bf_pk(acc0[r] + bias0, acc1[r] + bias1);
      if (m < 2) {
        __bf16* out = (m == 0) ? q : k;
        size_t base = ((size_t)bh * SEQP + srow) * DHEAD;
        out[base + col]      = bf_lo(u);
        out[base + 16 + col] = bf_hi(u);
      } else {
        vT[((size_t)bh * DHEAD + col)      * SEQP + srow] = bf_lo(u);
        vT[((size_t)bh * DHEAD + 16 + col) * SEQP + srow] = bf_hi(u);
      }
    }
  }
}

// Fused attention: scores (14 WMMA) -> register softmax (half-wave shfl_xor
// reductions, t>=197 masked) -> bf16 strip to LDS -> att@V (14 WMMA) -> residual.
// Never materializes the S x S attention matrix in HBM.
__global__ void vit_attention(const __bf16* __restrict__ q, const __bf16* __restrict__ k,
                              const __bf16* __restrict__ vT,
                              const float* __restrict__ xin, float* __restrict__ xout) {
  __shared__ __bf16 lds_att[16 * SEQP];
  const int lane = threadIdx.x;
  int idx = blockIdx.x;
  const int mt = idx % (SEQP / 16); idx /= (SEQP / 16);
  const int h  = idx % NHEADS;      idx /= NHEADS;
  const int b  = idx;
  const int bh = b * NHEADS + h;

  const __bf16* qp  = q  + (size_t)bh * SEQP * DHEAD;
  const __bf16* kp  = k  + (size_t)bh * SEQP * DHEAD;
  const __bf16* vTp = vT + (size_t)bh * DHEAD * SEQP;

  // scores: 16 x 224 strip, K = DHEAD = 32 (single WMMA step per tile)
  v16bf a = load_tile16(qp, DHEAD, mt * 16, 0, lane);
  v8f sc[SEQP / 16];
#pragma unroll
  for (int nt = 0; nt < SEQP / 16; ++nt) {
    v8f z = {};
    v16bf bt = load_tile16(kp, DHEAD, nt * 16, 0, lane);   // row-major K == B^T layout
    WMMA(z, a, bt);
    sc[nt] = z;
  }

  const float scale = 0.17677669529663687f;   // 1/sqrt(32)
  const int col = lane & 15;
  // softmax per row: C layout puts row M in lane-half 0 and M+8 in half 1,
  // so reductions use xor masks 8..1 only (stay within 16-lane halves).
#pragma unroll
  for (int r = 0; r < 8; ++r) {
    float mx = -1e30f;
#pragma unroll
    for (int nt = 0; nt < SEQP / 16; ++nt) {
      float vv = sc[nt][r] * scale;
      sc[nt][r] = vv;
      if (nt * 16 + col < SEQ) mx = fmaxf(mx, vv);
    }
#pragma unroll
    for (int o = 8; o >= 1; o >>= 1) mx = fmaxf(mx, __shfl_xor(mx, o, 32));
    float sum = 0.f;
#pragma unroll
    for (int nt = 0; nt < SEQP / 16; ++nt) {
      float ev = (nt * 16 + col < SEQ) ? __expf(sc[nt][r] - mx) : 0.f;
      sc[nt][r] = ev;
      sum += ev;
    }
#pragma unroll
    for (int o = 8; o >= 1; o >>= 1) sum += __shfl_xor(sum, o, 32);
    float inv = 1.f / sum;
#pragma unroll
    for (int nt = 0; nt < SEQP / 16; ++nt) sc[nt][r] *= inv;
  }

  // C-layout registers -> row-major bf16 strip in LDS (16 x 224)
  const int rbase = 8 * (lane >> 4);
#pragma unroll
  for (int nt = 0; nt < SEQP / 16; ++nt)
#pragma unroll
    for (int r = 0; r < 8; r += 2) {
      unsigned u = f2bf_pk(sc[nt][r], sc[nt][r + 1]);
      lds_att[(rbase + r)     * SEQP + nt * 16 + col] = bf_lo(u);
      lds_att[(rbase + r + 1) * SEQP + nt * 16 + col] = bf_hi(u);
    }
  __syncthreads();

  // o = att @ V : K loop 224 in 7 steps of 32; A tiles from LDS (2x ds_load_b128),
  // B tiles from transposed V (contiguous-K vector loads).
  v8f acc0 = {}, acc1 = {};
  const int hh = lane >> 4;
#pragma unroll
  for (int kk = 0; kk < SEQP; kk += 32) {
    const int abase = (lane & 15) * SEQP + kk;
    v8bf lo = *(const v8bf*)&lds_att[abase + hh * 8];
    v8bf hi = *(const v8bf*)&lds_att[abase + 16 + hh * 8];
    v16bf aa;
#pragma unroll
    for (int e = 0; e < 8; ++e) { aa[e] = lo[e]; aa[8 + e] = hi[e]; }
    v16bf b0 = load_tile16(vTp, SEQP, 0, kk, lane);
    v16bf b1 = load_tile16(vTp, SEQP, 16, kk, lane);
    WMMA(acc0, aa, b0);
    WMMA(acc1, aa, b1);
  }
#pragma unroll
  for (int r = 0; r < 8; ++r) {
    int srow = mt * 16 + r + rbase;
    size_t base = ((size_t)b * SEQP + srow) * DIM + h * DHEAD;
    xout[base + col]      = xin[base + col]      + acc0[r];
    xout[base + 16 + col] = xin[base + 16 + col] + acc1[r];
  }
}

// MLP up-proj + exact GELU: (B*224, 64) x (64, 256). 4 waves/block, 4 N-tiles each.
__global__ void vit_mlp1(const __bf16* __restrict__ m, const __bf16* __restrict__ W1b,
                         const float* __restrict__ b1, __bf16* __restrict__ hmid) {
  const int lane  = threadIdx.x & 31;
  const int wave  = threadIdx.x >> 5;
  const int mbase = blockIdx.x * 16;
  v8f acc[4] = {};
#pragma unroll
  for (int kk = 0; kk < DIM; kk += 32) {
    v16bf a = load_tile16(m, DIM, mbase, kk, lane);
#pragma unroll
    for (int j = 0; j < 4; ++j) {
      v16bf bt = load_b_packed(W1b, 16, kk >> 5, wave * 4 + j, lane);
      WMMA(acc[j], a, bt);
    }
  }
  const int col = lane & 15;
#pragma unroll
  for (int j = 0; j < 4; ++j) {
    int n = (wave * 4 + j) * 16 + col;
    float bias = b1[n];
#pragma unroll
    for (int r = 0; r < 8; r += 2) {
      int row = mbase + r + 8 * (lane >> 4);
      float x0 = acc[j][r] + bias;
      float x1 = acc[j][r + 1] + bias;
      float g0 = 0.5f * x0 * (1.f + erff(x0 * 0.70710678118654752f));
      float g1 = 0.5f * x1 * (1.f + erff(x1 * 0.70710678118654752f));
      unsigned u = f2bf_pk(g0, g1);
      hmid[(size_t)row * (4 * DIM) + n]       = bf_lo(u);
      hmid[(size_t)(row + 1) * (4 * DIM) + n] = bf_hi(u);
    }
  }
}

// MLP down-proj + residual: (B*224, 256) x (256, 64). One wave per 16x64 tile.
__global__ void vit_mlp2(const __bf16* __restrict__ hmid, const __bf16* __restrict__ W2b,
                         const float* __restrict__ b2, const float* __restrict__ xres,
                         float* __restrict__ xo) {
  const int lane  = threadIdx.x;
  const int mbase = blockIdx.x * 16;
  v8f acc[4] = {};
#pragma unroll
  for (int kk = 0; kk < 4 * DIM; kk += 32) {
    v16bf a = load_tile16(hmid, 4 * DIM, mbase, kk, lane);
#pragma unroll
    for (int j = 0; j < 4; ++j) {
      v16bf bt = load_b_packed(W2b, 4, kk >> 5, j, lane);
      WMMA(acc[j], a, bt);
    }
  }
  const int col = lane & 15;
#pragma unroll
  for (int j = 0; j < 4; ++j) {
    int n = j * 16 + col;
    float bias = b2[n];
#pragma unroll
    for (int r = 0; r < 8; ++r) {
      int row = mbase + r + 8 * (lane >> 4);
      xo[(size_t)row * DIM + n] = xres[(size_t)row * DIM + n] + acc[j][r] + bias;
    }
  }
}

// Head GEMM: cls rows (256, 64) x (64, 1008 padded). A converted fp32->bf16
// on the fly via float4 loads + packed conversion.
__global__ void vit_head(const float* __restrict__ xA, const __bf16* __restrict__ hWb,
                         const float* __restrict__ hb, float* __restrict__ logits) {
  const int lane = threadIdx.x;
  const int mt = blockIdx.x & 15;
  const int nt = blockIdx.x >> 4;
  const int h  = lane >> 4;
  v8f acc = {};
  const int brow = mt * 16 + (lane & 15);
  const float* xr = xA + (size_t)brow * SEQP * DIM;   // row s=0 (cls)
#pragma unroll
  for (int kk = 0; kk < DIM; kk += 32) {
    union { v16bf v; unsigned u[8]; } a;
#pragma unroll
    for (int run = 0; run < 2; ++run) {
      const float* pf = xr + kk + run * 16 + h * 8;
      float4 f0 = *(const float4*)pf;
      float4 f1 = *(const float4*)(pf + 4);
      a.u[run * 4 + 0] = f2bf_pk(f0.x, f0.y);
      a.u[run * 4 + 1] = f2bf_pk(f0.z, f0.w);
      a.u[run * 4 + 2] = f2bf_pk(f1.x, f1.y);
      a.u[run * 4 + 3] = f2bf_pk(f1.z, f1.w);
    }
    v16bf bt = load_b_packed(hWb, NCLSP / 16, kk >> 5, nt, lane);
    WMMA(acc, a.v, bt);
  }
  const int col = lane & 15;
  const int n = nt * 16 + col;
  float bias = (n < NCLS) ? hb[n] : 0.f;
#pragma unroll
  for (int r = 0; r < 8; ++r) {
    int bb = mt * 16 + r + 8 * h;
    logits[(size_t)bb * NCLSP + n] = acc[r] + bias;
  }
}

__global__ void vit_head_softmax(const float* __restrict__ logits, float* __restrict__ out) {
  const int lane = threadIdx.x;
  const int b = blockIdx.x;
  const float* lr = logits + (size_t)b * NCLSP;
  float mx = -1e30f;
  for (int t = lane; t < NCLS; t += 32) mx = fmaxf(mx, lr[t]);
#pragma unroll
  for (int o = 16; o >= 1; o >>= 1) mx = fmaxf(mx, __shfl_xor(mx, o, 32));
  float sum = 0.f;
  for (int t = lane; t < NCLS; t += 32) sum += __expf(lr[t] - mx);
#pragma unroll
  for (int o = 16; o >= 1; o >>= 1) sum += __shfl_xor(sum, o, 32);
  float inv = 1.f / sum;
  for (int t = lane; t < NCLS; t += 32) out[(size_t)b * NCLS + t] = __expf(lr[t] - mx) * inv;
}

// ------------------------------- host --------------------------------------

extern "C" void kernel_launch(void* const* d_in, const int* in_sizes, int n_in,
                              void* d_out, int out_size, void* d_ws, size_t ws_size,
                              hipStream_t stream) {
  (void)in_sizes; (void)n_in; (void)out_size; (void)ws_size;

  const float* images  = (const float*)d_in[0];
  const float* W_patch = (const float*)d_in[1];
  const float* b_patch = (const float*)d_in[2];
  const float* cls_tok = (const float*)d_in[3];
  const float* pos     = (const float*)d_in[4];
  const float* ln1_g   = (const float*)d_in[5];
  const float* ln1_b   = (const float*)d_in[6];
  const float* ln2_g   = (const float*)d_in[7];
  const float* ln2_b   = (const float*)d_in[8];
  const float* qW      = (const float*)d_in[9];
  const float* qb      = (const float*)d_in[10];
  const float* kW      = (const float*)d_in[11];
  const float* kb      = (const float*)d_in[12];
  const float* vW      = (const float*)d_in[13];
  const float* vb      = (const float*)d_in[14];
  const float* W1      = (const float*)d_in[15];
  const float* b1      = (const float*)d_in[16];
  const float* W2      = (const float*)d_in[17];
  const float* b2      = (const float*)d_in[18];
  const float* hW      = (const float*)d_in[19];
  const float* hb      = (const float*)d_in[20];

  char* ws = (char*)d_ws;
  size_t off = 0;
  auto salloc = [&](size_t bytes) -> void* {
    void* p = ws + off;
    off = (off + bytes + 255) & ~(size_t)255;
    return p;
  };

  const size_t NACT = (size_t)BATCH * SEQP * DIM;           // 3,670,016 elements
  __bf16* wpatch_bf = (__bf16*)salloc((size_t)DPATCH * DIM * 2);
  __bf16* qW_bf     = (__bf16*)salloc((size_t)NLAYER * NHEADS * DHEAD * DHEAD * 2);
  __bf16* kW_bf     = (__bf16*)salloc((size_t)NLAYER * NHEADS * DHEAD * DHEAD * 2);
  __bf16* vW_bf     = (__bf16*)salloc((size_t)NLAYER * NHEADS * DHEAD * DHEAD * 2);
  __bf16* W1_bf     = (__bf16*)salloc((size_t)NLAYER * DIM * 4 * DIM * 2);
  __bf16* W2_bf     = (__bf16*)salloc((size_t)NLAYER * 4 * DIM * DIM * 2);
  __bf16* hW_bf     = (__bf16*)salloc((size_t)DIM * NCLSP * 2);
  float*  xA        = (float*)salloc(NACT * 4);
  float*  xB        = (float*)salloc(NACT * 4);
  __bf16* ybuf      = (__bf16*)salloc(NACT * 2);
  __bf16* qbuf      = (__bf16*)salloc(NACT * 2);
  __bf16* kbuf      = (__bf16*)salloc(NACT * 2);
  __bf16* vTbuf     = (__bf16*)salloc(NACT * 2);
  __bf16* hmid      = (__bf16*)salloc((size_t)BATCH * SEQP * 4 * DIM * 2);
  float*  logits    = (float*)salloc((size_t)BATCH * NCLSP * 4);

  auto pack = [&](const float* s, __bf16* d, int K, int N, int Nsrc, int nmat) {
    int total = K * N * nmat;
    vit_pack_b<<<(total + 255) / 256, 256, 0, stream>>>(s, d, K, N, Nsrc, nmat);
  };
  pack(W_patch, wpatch_bf, DPATCH, DIM, DIM, 1);
  pack(qW, qW_bf, DHEAD, DHEAD, DHEAD, NLAYER * NHEADS);
  pack(kW, kW_bf, DHEAD, DHEAD, DHEAD, NLAYER * NHEADS);
  pack(vW, vW_bf, DHEAD, DHEAD, DHEAD, NLAYER * NHEADS);
  pack(W1, W1_bf, DIM, 4 * DIM, 4 * DIM, NLAYER);
  pack(W2, W2_bf, 4 * DIM, DIM, DIM, NLAYER);
  pack(hW, hW_bf, DIM, NCLSP, NCLS, 1);

  vit_patch_embed<<<BATCH * NPATCH / 16, 32, 0, stream>>>(images, wpatch_bf, b_patch, pos, xA);
  vit_init_rows<<<BATCH, 64, 0, stream>>>(cls_tok, pos, xA);

  const int ROWS = BATCH * SEQP;
  const int BHTS = BATCH * NHEADS * (SEQP / 16);
  for (int l = 0; l < NLAYER; ++l) {
    vit_layernorm<<<ROWS, 32, 0, stream>>>(xA, ln1_g + l * DIM, ln1_b + l * DIM, ybuf);
    vit_qkv<<<BHTS, 32, 0, stream>>>(ybuf,
        qW_bf + (size_t)l * NHEADS * DHEAD * DHEAD,
        kW_bf + (size_t)l * NHEADS * DHEAD * DHEAD,
        vW_bf + (size_t)l * NHEADS * DHEAD * DHEAD,
        qb + l * DIM, kb + l * DIM, vb + l * DIM,
        qbuf, kbuf, vTbuf);
    vit_attention<<<BHTS, 32, 0, stream>>>(qbuf, kbuf, vTbuf, xA, xB);
    vit_layernorm<<<ROWS, 32, 0, stream>>>(xB, ln2_g + l * DIM, ln2_b + l * DIM, ybuf);
    vit_mlp1<<<ROWS / 16, 128, 0, stream>>>(ybuf, W1_bf + (size_t)l * DIM * 4 * DIM,
                                            b1 + l * 4 * DIM, hmid);
    vit_mlp2<<<ROWS / 16, 32, 0, stream>>>(hmid, W2_bf + (size_t)l * 4 * DIM * DIM,
                                           b2 + l * DIM, xB, xA);
  }

  vit_head<<<(BATCH / 16) * (NCLSP / 16), 32, 0, stream>>>(xA, hW_bf, hb, logits);
  vit_head_softmax<<<BATCH, 32, 0, stream>>>(logits, (float*)d_out);
}
